// FastSpeech2WithMask_23974507446351
// MI455X (gfx1250) — compile-verified, matrix operand order checked by reading, not run on hardware
//
#include <hip/hip_runtime.h>
#include <hip/hip_fp16.h>

// ---------------------------------------------------------------------------
// FastSpeech2-with-mask forward for gfx1250 (MI455X).
// All GEMMs / attention matmuls run on v_wmma_f32_16x16x32_f16 (wave32 WMMA).
// Attention is flash-style (streaming kv tiles, online softmax) so no O(S^2)
// energy buffer is materialized.
// ---------------------------------------------------------------------------

typedef _Float16 h16;
typedef _Float16 v16h __attribute__((ext_vector_type(16)));
typedef _Float16 v8h  __attribute__((ext_vector_type(8)));
typedef float    v8f  __attribute__((ext_vector_type(8)));

#define WB   8
#define WT   512
#define WD   512
#define WH   2
#define WDK  256
#define NEGV (-1.0e10f)

__device__ __forceinline__ v8f zero8() {
  v8f z;
#pragma unroll
  for (int i = 0; i < 8; ++i) z[i] = 0.0f;
  return z;
}

// Load one WMMA A/B operand chunk (16x32 / 32x16, 16-bit) for this lane.
// Lanes 0-15 hold row/col (lane&15) with K in {k..k+7, k+16..k+23} where
// k = chunk_base; lanes 16-31 use k+8 (per CDNA5 ISA 7.12.2 layout).
__device__ __forceinline__ v16h load_ab(const h16* row, int k0) {
  v8h lo = *(const v8h*)(row + k0);
  v8h hi = *(const v8h*)(row + k0 + 16);
  v16h r;
#pragma unroll
  for (int i = 0; i < 8; ++i) { r[i] = lo[i]; r[i + 8] = hi[i]; }
  return r;
}

__device__ __forceinline__ v8f wmma16(v16h a, v16h b, v8f c) {
  // D = A(16x32 f16) * B(32x16 f16) + C(16x16 f32)
  return __builtin_amdgcn_wmma_f32_16x16x32_f16(
      false, a, false, b, (short)0, c, false, false);
}

// ---------------------------------------------------------------------------
// Weight f32 -> f16 conversion (wq | wk | wv | fc_w packed contiguously).
// ---------------------------------------------------------------------------
__global__ void k_cvt_w(const float* wq, const float* wk, const float* wv,
                        const float* fw, h16* out) {
  int i = blockIdx.x * 256 + threadIdx.x;
  const int DD = WD * WD;
  if (i < DD)            out[i] = (h16)wq[i];
  else if (i < 2 * DD)   out[i] = (h16)wk[i - DD];
  else if (i < 3 * DD)   out[i] = (h16)wv[i - 2 * DD];
  else if (i < 4 * DD)   out[i] = (h16)fw[i - 3 * DD];
}

// ---------------------------------------------------------------------------
// Embedding gather -> f16 activations; rows >= T zero-filled (pad tiles).
// grid.x = B*SP blocks, 256 threads, 2 elems/thread (D=512).
// ---------------------------------------------------------------------------
__global__ void k_embed(const int* __restrict__ text,
                        const float* __restrict__ emb,
                        h16* __restrict__ xa, int SP) {
  int bt = blockIdx.x;
  int b = bt / SP, t = bt % SP;
  int d = threadIdx.x;
  h16* row = xa + (size_t)bt * WD;
  if (t < WT) {
    int tok = text[b * WT + t];
    const float* e = emb + (size_t)tok * WD;
    row[d]       = (h16)e[d];
    row[d + 256] = (h16)e[d + 256];
  } else {
    row[d]       = (h16)0.0f;
    row[d + 256] = (h16)0.0f;
  }
}

// ---------------------------------------------------------------------------
// WMMA GEMM: out[m,n] = bias[n] + sum_k A[m,k] * W[n,k]   (K = D = 512)
// A rows live at (b*SP + t)*D. 128 threads = 4 waves, block tile 32x128.
// hmode 1: f16 store as Q/K layout  [b,h,t,dk]
// hmode 2: f16 store as V^T layout  [b,h,dk,t]
// outF   : f32 store (b*frows + t)*D + n, guarded t < svalid (fc outputs)
// ---------------------------------------------------------------------------
__global__ void k_gemm(const h16* __restrict__ A, const h16* __restrict__ W,
                       const float* __restrict__ bias,
                       h16* __restrict__ outH, int hmode,
                       float* __restrict__ outF, int frows,
                       int SP, int svalid, int mtiles) {
  int b    = blockIdx.x / mtiles;
  int mt   = blockIdx.x % mtiles;
  int wid  = threadIdx.x >> 5;
  int lane = threadIdx.x & 31;
  int mw = wid >> 1, nw = wid & 1;
  int t0    = mt * 32 + mw * 16;
  int nbase = blockIdx.y * 128 + nw * 64;
  int lc    = lane & 15;
  int k0    = (lane >> 4) ? 8 : 0;
  int rbase = (lane >> 4) ? 8 : 0;

  const h16* arow = A + ((size_t)b * SP + t0 + lc) * WD;

  v8f acc[4];
#pragma unroll
  for (int j = 0; j < 4; ++j) acc[j] = zero8();

#pragma unroll 4
  for (int c = 0; c < 16; ++c) {
    v16h a = load_ab(arow, c * 32 + k0);
#pragma unroll
    for (int j = 0; j < 4; ++j) {
      const h16* wrow = W + (size_t)(nbase + j * 16 + lc) * WD;
      v16h bb = load_ab(wrow, c * 32 + k0);
      acc[j] = wmma16(a, bb, acc[j]);
    }
  }

#pragma unroll
  for (int j = 0; j < 4; ++j) {
    int col = nbase + j * 16 + lc;
    float bval = bias[col];
    int hh = col / WDK, dk = col % WDK;
#pragma unroll
    for (int r = 0; r < 8; ++r) {
      int t = t0 + rbase + r;
      float v = acc[j][r] + bval;
      if (outH) {
        size_t idx = (hmode == 1)
            ? (((size_t)(b * WH + hh) * SP + t) * WDK + dk)
            : (((size_t)(b * WH + hh) * WDK + dk) * SP + t);
        outH[idx] = (h16)v;
      }
      if (outF && t < svalid)
        outF[((size_t)b * frows + t) * WD + col] = v;
    }
  }
}

// ---------------------------------------------------------------------------
// Flash attention (self-attn). Block = 4 waves: (q-subtile, d-half).
// Each wave: 16 q rows x 128 output dims, streams kv in tiles of 32.
// S   : softmax axis length (T or mel_max); kv >= S excluded via -inf.
// len : valid length per batch; invalid positions get NEG (matches ref).
// ---------------------------------------------------------------------------
__global__ void k_flash(const h16* __restrict__ Q, const h16* __restrict__ K,
                        const h16* __restrict__ VT, h16* __restrict__ O,
                        const int* __restrict__ lens, int SP, int S) {
  __shared__ __align__(16) h16 plds[4][16][32];
  int b = blockIdx.z, hh = blockIdx.y;
  int wid  = threadIdx.x >> 5;
  int lane = threadIdx.x & 31;
  int qw = wid >> 1, dw = wid & 1;
  int qbase = blockIdx.x * 32 + qw * 16;
  int lc    = lane & 15;
  int k0    = (lane >> 4) ? 8 : 0;
  int rbase = (lane >> 4) ? 8 : 0;
  int len   = lens[b];
  const size_t head = (size_t)(b * WH + hh);

  // Preload this wave's 16 q rows (DK=256 -> 8 chunks of 32).
  const h16* qrow = Q + (head * SP + qbase + lc) * WDK;
  v16h qa[8];
#pragma unroll
  for (int c = 0; c < 8; ++c) qa[c] = load_ab(qrow, c * 32 + k0);

  v8f acc[8];
#pragma unroll
  for (int j = 0; j < 8; ++j) acc[j] = zero8();
  float rm[8], rl[8];
#pragma unroll
  for (int r = 0; r < 8; ++r) { rm[r] = -__builtin_inff(); rl[r] = 0.0f; }

  const float scale = 0.0625f;  // 1/sqrt(DK=256)
  int nkv = (S + 31) / 32;
  for (int kt = 0; kt < nkv; ++kt) {
    int kt0 = kt * 32;

    // S-tile: q(16x256) @ k^T(256x32) -> two 16x16 f32 tiles.
    v8f s0 = zero8(), s1 = zero8();
    const h16* krow0 = K + (head * SP + kt0 + lc) * WDK;
    const h16* krow1 = krow0 + (size_t)16 * WDK;
#pragma unroll
    for (int c = 0; c < 8; ++c) {
      v16h b0 = load_ab(krow0, c * 32 + k0);
      v16h b1 = load_ab(krow1, c * 32 + k0);
      s0 = wmma16(qa[c], b0, s0);
      s1 = wmma16(qa[c], b1, s1);
    }

    int col0 = kt0 + lc, col1 = col0 + 16;
    bool kv0 = col0 < len, kv1 = col1 < len;
#pragma unroll
    for (int r = 0; r < 8; ++r) {
      int qidx = qbase + rbase + r;
      bool qv = qidx < len;
      float e0 = (qv && kv0) ? s0[r] * scale : NEGV;
      float e1 = (qv && kv1) ? s1[r] * scale : NEGV;
      e0 = (col0 < S) ? e0 : -__builtin_inff();
      e1 = (col1 < S) ? e1 : -__builtin_inff();

      float mx = fmaxf(e0, e1);
      mx = fmaxf(mx, __shfl_xor(mx, 1, 16));
      mx = fmaxf(mx, __shfl_xor(mx, 2, 16));
      mx = fmaxf(mx, __shfl_xor(mx, 4, 16));
      mx = fmaxf(mx, __shfl_xor(mx, 8, 16));

      float mnew  = fmaxf(rm[r], mx);
      float alpha = __expf(rm[r] - mnew);
      float p0 = __expf(e0 - mnew);
      float p1 = __expf(e1 - mnew);
      float rs = p0 + p1;
      rs += __shfl_xor(rs, 1, 16);
      rs += __shfl_xor(rs, 2, 16);
      rs += __shfl_xor(rs, 4, 16);
      rs += __shfl_xor(rs, 8, 16);
      rl[r] = rl[r] * alpha + rs;
      rm[r] = mnew;

      // C layout -> LDS so P can be reloaded in A-operand layout.
      plds[wid][rbase + r][lc]      = (h16)p0;
      plds[wid][rbase + r][lc + 16] = (h16)p1;
#pragma unroll
      for (int j = 0; j < 8; ++j) acc[j][r] *= alpha;
    }

    // P(16x32) @ V(32x128-slice): 8 WMMAs. V^T rows are contiguous in t.
    v16h pa = load_ab(&plds[wid][lc][0], k0);
#pragma unroll
    for (int j = 0; j < 8; ++j) {
      const h16* vrow = VT + (head * WDK + dw * 128 + j * 16 + lc) * SP;
      v16h bv = load_ab(vrow + kt0, k0);
      acc[j] = wmma16(pa, bv, acc[j]);
    }
  }

  // Normalize and store (heads concatenated along D).
#pragma unroll
  for (int j = 0; j < 8; ++j) {
#pragma unroll
    for (int r = 0; r < 8; ++r) {
      int t   = qbase + rbase + r;
      int col = hh * WDK + dw * 128 + j * 16 + lc;
      O[((size_t)b * SP + t) * WD + col] = (h16)(acc[j][r] / rl[r]);
    }
  }
}

// ---------------------------------------------------------------------------
// Per-batch duration cumsum (T=512, trivial serial scan per batch).
// ---------------------------------------------------------------------------
__global__ void k_cumsum(const float* __restrict__ dur, int* __restrict__ cum) {
  int b = threadIdx.x;
  if (b >= WB) return;
  int s = 0;
  for (int i = 0; i < WT; ++i) {
    s += (int)dur[b * WT + i];
    cum[b * WT + i] = s;
  }
}

// ---------------------------------------------------------------------------
// Length regulate: searchsorted(cum, t, 'right'), gather row, apply validity.
// Also writes mel_mask into d_out tail. grid = (SP, B), 256 threads.
// ---------------------------------------------------------------------------
__global__ void k_regulate(const float* __restrict__ x1,
                           const int* __restrict__ cum,
                           const int* __restrict__ mlen,
                           h16* __restrict__ mel,
                           float* __restrict__ mask_out,
                           int mel_max, int SP) {
  int t = blockIdx.x, b = blockIdx.y;
  int d = threadIdx.x;
  h16* row = mel + ((size_t)b * SP + t) * WD;
  if (t >= mel_max) {
    row[d] = (h16)0.0f;
    row[d + 256] = (h16)0.0f;
    return;
  }
  const int* c = cum + b * WT;
  int lo = 0, hi = WT;
  while (lo < hi) {
    int mid = (lo + hi) >> 1;
    if (c[mid] <= t) lo = mid + 1; else hi = mid;
  }
  int idx = lo < (WT - 1) ? lo : (WT - 1);
  bool valid = t < mlen[b];
  const float* src = x1 + ((size_t)b * WT + idx) * WD;
  row[d]       = valid ? (h16)src[d]       : (h16)0.0f;
  row[d + 256] = valid ? (h16)src[d + 256] : (h16)0.0f;
  if (d == 0) mask_out[(size_t)b * mel_max + t] = valid ? 1.0f : 0.0f;
}

// ---------------------------------------------------------------------------
// Host orchestration.
// ---------------------------------------------------------------------------
extern "C" void kernel_launch(void* const* d_in, const int* in_sizes, int n_in,
                              void* d_out, int out_size, void* d_ws,
                              size_t ws_size, hipStream_t stream) {
  (void)in_sizes; (void)n_in; (void)ws_size;
  const int B = WB, T = WT, D = WD, H = WH;

  const int*   text     = (const int*)d_in[0];
  const float* duration = (const float*)d_in[1];
  const int*   text_len = (const int*)d_in[2];
  const int*   mel_len  = (const int*)d_in[3];
  // d_in[4] = mel_max scalar on device; derived from out_size instead.
  const float* emb = (const float*)d_in[5];
  const float* wq  = (const float*)d_in[6];
  const float* bq  = (const float*)d_in[7];
  const float* wk  = (const float*)d_in[8];
  const float* bk  = (const float*)d_in[9];
  const float* wv  = (const float*)d_in[10];
  const float* bv  = (const float*)d_in[11];
  const float* fw  = (const float*)d_in[12];
  const float* fb  = (const float*)d_in[13];

  // out = (B, mel_max, D) f32  ++  mel_mask = (B, mel_max) f32
  int mel_max = out_size / (B * (D + 1));
  int SP = (mel_max + 31) & ~31;   // padded mel length (>= T, durations >= 1)
  if (SP < T) SP = T;

  char* ws = (char*)d_ws;
  size_t off = 0;
  const size_t DD = (size_t)D * D;
  const size_t ACT = (size_t)B * SP * D;   // f16 activation buffer elems
  h16* wf16 = (h16*)(ws + off); off += 4 * DD * sizeof(h16);
  h16* xa   = (h16*)(ws + off); off += ACT * sizeof(h16);  // x / mel repr
  h16* qb   = (h16*)(ws + off); off += ACT * sizeof(h16);
  h16* kb   = (h16*)(ws + off); off += ACT * sizeof(h16);
  h16* vtb  = (h16*)(ws + off); off += ACT * sizeof(h16);
  h16* ob   = (h16*)(ws + off); off += ACT * sizeof(h16);
  float* x1 = (float*)(ws + off); off += (size_t)B * T * D * sizeof(float);
  int* cum  = (int*)(ws + off);   off += (size_t)B * T * sizeof(int);

  float* outp     = (float*)d_out;
  float* mask_out = outp + (size_t)B * mel_max * D;

  // Stage 0: weight conversion + embedding.
  k_cvt_w<<<(int)((4 * DD + 255) / 256), 256, 0, stream>>>(wq, wk, wv, fw, wf16);
  k_embed<<<B * SP, 256, 0, stream>>>(text, emb, xa, SP);

  // Stage 1: text MHA.
  int mt1 = T / 32;
  dim3 g1(B * mt1, D / 128);
  k_gemm<<<g1, 128, 0, stream>>>(xa, wf16,          bq, qb,  1, nullptr, 0, SP, T, mt1);
  k_gemm<<<g1, 128, 0, stream>>>(xa, wf16 + DD,     bk, kb,  1, nullptr, 0, SP, T, mt1);
  k_gemm<<<g1, 128, 0, stream>>>(xa, wf16 + 2 * DD, bv, vtb, 2, nullptr, 0, SP, T, mt1);
  k_flash<<<dim3(mt1, H, B), 128, 0, stream>>>(qb, kb, vtb, ob, text_len, SP, T);
  k_gemm<<<g1, 128, 0, stream>>>(ob, wf16 + 3 * DD, fb, nullptr, 0, x1, T, SP, T, mt1);

  // Length regulation.
  k_cumsum<<<1, B, 0, stream>>>(duration, cum);
  k_regulate<<<dim3(SP, B), 256, 0, stream>>>(x1, cum, mel_len, xa, mask_out,
                                              mel_max, SP);

  // Stage 2: mel MHA (same weights), output -> d_out.
  int mt2 = SP / 32;
  dim3 g2(B * mt2, D / 128);
  k_gemm<<<g2, 128, 0, stream>>>(xa, wf16,          bq, qb,  1, nullptr, 0, SP, mel_max, mt2);
  k_gemm<<<g2, 128, 0, stream>>>(xa, wf16 + DD,     bk, kb,  1, nullptr, 0, SP, mel_max, mt2);
  k_gemm<<<g2, 128, 0, stream>>>(xa, wf16 + 2 * DD, bv, vtb, 2, nullptr, 0, SP, mel_max, mt2);
  k_flash<<<dim3(mt2, H, B), 128, 0, stream>>>(qb, kb, vtb, ob, mel_len, SP, mel_max);
  k_gemm<<<g2, 128, 0, stream>>>(ob, wf16 + 3 * DD, fb, nullptr, 0, outp, mel_max,
                                 SP, mel_max, mt2);
}
